// RotaryPositionalMultiHeadAttention_87282325389403
// MI455X (gfx1250) — matrix operand order repaired and checked by reading, not compiled, not run on hardware
//
#include <hip/hip_runtime.h>
#include <hip/hip_bf16.h>

// RotaryPositionalMultiHeadAttention forward for MI455X (gfx1250, wave32, WMMA).
// f32->bf16 convert -> WMMA GEMM projections (async LDS tiles) -> RoPE repack ->
// flash-attention with v_wmma_f32_16x16x32_bf16 + LDS-staged K/V -> WMMA out proj.

#define BDIM   2
#define SEQ    2048
#define DMODEL 1024
#define NHEADS 16
#define DKH    64
#define BSROWS (BDIM * SEQ)   // 4096

typedef __attribute__((ext_vector_type(16))) __bf16 bf16x16;
typedef __attribute__((ext_vector_type(8)))  float  f32x8;
typedef __attribute__((ext_vector_type(4)))  int    v4i;

union BFrag {
  bf16x16 v;
  unsigned short h[16];
  unsigned int   u[8];
};

union U128 {
  uint4 q;
  unsigned short s[8];
};

__device__ __forceinline__ unsigned short f2bf(float f) {
  union { float f; unsigned u; } c; c.f = f;
  unsigned r = c.u + 0x7FFFu + ((c.u >> 16) & 1u);   // round-to-nearest-even
  return (unsigned short)(r >> 16);
}

// CDNA5 16-bit A-matrix (16x32) per-lane K offsets: VGPR v<4 -> K=2v,2v+1 ;
// v>=4 -> K=16+2(v-4) ; lanes 16..31 shift K by +8.
__device__ __forceinline__ int a_koff(int v, int half) {
  return (v < 4 ? 2 * v : 16 + 2 * (v - 4)) + 8 * half;
}
// CDNA5 16-bit B-matrix (32x16): same K order as A: VGPR v<4 -> K=2v,2v+1 ;
// v>=4 -> K=16+2(v-4) ; lanes 16..31 shift K by +8.
__device__ __forceinline__ int b_koff(int v, int half) {
  return (v < 4 ? 2 * v : 16 + 2 * (v - 4)) + 8 * half;
}

// ---------------------------------------------------------------------------
// gfx1250 async global->LDS copy (ASYNCcnt-tracked), with sync fallback.
// Builtin signature (from hipcc diagnostic): (v4i AS1*, v4i AS3*, imm, imm).
// ---------------------------------------------------------------------------
#if defined(__AMDGCN__) && __has_builtin(__builtin_amdgcn_global_load_async_to_lds_b128)
#define RPMHA_ASYNC_LDS 1
#else
#define RPMHA_ASYNC_LDS 0
#endif

__device__ __forceinline__ void rpmha_copy16(const unsigned short* __restrict__ g,
                                             unsigned short* l) {
#if RPMHA_ASYNC_LDS
  __builtin_amdgcn_global_load_async_to_lds_b128(
      (__attribute__((address_space(1))) v4i*)(size_t)g,
      (__attribute__((address_space(3))) v4i*)(size_t)l, 0, 0);
#else
  *(uint4*)l = *(const uint4*)g;
#endif
}

__device__ __forceinline__ void rpmha_wait_async() {
#if RPMHA_ASYNC_LDS
#if __has_builtin(__builtin_amdgcn_s_wait_asynccnt)
  __builtin_amdgcn_s_wait_asynccnt(0);
#else
  asm volatile("s_wait_asynccnt 0x0" ::: "memory");
#endif
#endif
}

// ---------------------------------------------------------------------------
// fp32 -> bf16 convert
// ---------------------------------------------------------------------------
__global__ void rpmha_f32_to_bf16(const float* __restrict__ x,
                                  unsigned short* __restrict__ y, int n) {
  int i = blockIdx.x * blockDim.x + threadIdx.x;
  if (i < n) y[i] = f2bf(x[i]);
}

// ---------------------------------------------------------------------------
// GEMM: C(MxN, f32) = A(MxK, bf16) * B(KxN, bf16) + bias(N)
// Block tile 128x64, 8 waves (4x2), each wave 32x32 via 2x2 WMMA tiles, K step 32.
// A tile staged row-major via async b128 copies; B tile staged TRANSPOSED so
// every B-fragment register is one aligned b32 LDS read (no b16 packing).
// ---------------------------------------------------------------------------
__global__ __launch_bounds__(256)
void rpmha_gemm_bf16(const unsigned short* __restrict__ A,
                     const unsigned short* __restrict__ Bm,
                     const float* __restrict__ bias,
                     float* __restrict__ C,
                     int M, int N, int K) {
  __shared__ unsigned short As[128][32];   // row-major, 16B-aligned rows
  __shared__ unsigned short BsT[64][34];   // transposed (n-major), padded

  const int m0 = blockIdx.y * 128;
  const int n0 = blockIdx.x * 64;
  const int t    = threadIdx.x;
  const int wave = t >> 5;
  const int lane = t & 31;
  const int half = lane >> 4;
  const int mrow = lane & 15;
  const int ncol = lane & 15;
  const int wm = wave & 3;      // 0..3  -> 32-row strip
  const int wn = wave >> 2;     // 0..1  -> 32-col strip

  f32x8 acc[2][2];
#pragma unroll
  for (int a = 0; a < 2; ++a)
#pragma unroll
    for (int b = 0; b < 2; ++b)
      acc[a][b] = (f32x8){0.f, 0.f, 0.f, 0.f, 0.f, 0.f, 0.f, 0.f};

  const int arow = t >> 1, acol = (t & 1) * 16;   // A: 16 elems / thread
  const int brow = t >> 3, bcol = (t & 7) * 8;    // B: 8 elems / thread

  for (int k0 = 0; k0 < K; k0 += 32) {
    {
      const unsigned short* gA = A + (size_t)(m0 + arow) * K + k0 + acol;
      rpmha_copy16(gA, &As[arow][acol]);
      rpmha_copy16(gA + 8, &As[arow][acol + 8]);

      U128 bv;
      bv.q = *(const uint4*)(Bm + (size_t)(k0 + brow) * N + n0 + bcol);
#pragma unroll
      for (int e = 0; e < 8; ++e) BsT[bcol + e][brow] = bv.s[e];
    }
    rpmha_wait_async();
    __syncthreads();

    BFrag aF[2], bF[2];
#pragma unroll
    for (int mi = 0; mi < 2; ++mi)
#pragma unroll
      for (int v = 0; v < 8; ++v)
        aF[mi].u[v] =
            *(const unsigned*)&As[wm * 32 + mi * 16 + mrow][a_koff(v, half)];
#pragma unroll
    for (int ni = 0; ni < 2; ++ni)
#pragma unroll
      for (int v = 0; v < 8; ++v)
        bF[ni].u[v] =
            *(const unsigned*)&BsT[wn * 32 + ni * 16 + ncol][b_koff(v, half)];
#pragma unroll
    for (int mi = 0; mi < 2; ++mi)
#pragma unroll
      for (int ni = 0; ni < 2; ++ni)
        acc[mi][ni] = __builtin_amdgcn_wmma_f32_16x16x32_bf16(
            false, aF[mi].v, false, bF[ni].v, (short)0, acc[mi][ni], false,
            false);
    __syncthreads();
  }

#pragma unroll
  for (int ni = 0; ni < 2; ++ni) {
    const int n = n0 + wn * 32 + ni * 16 + ncol;
    const float bv = bias[n];
#pragma unroll
    for (int mi = 0; mi < 2; ++mi)
#pragma unroll
      for (int i = 0; i < 8; ++i) {
        const int m = m0 + wm * 32 + mi * 16 + 8 * half + i;
        C[(size_t)m * N + n] = acc[mi][ni][i] + bv;
      }
  }
}

// ---------------------------------------------------------------------------
// RoPE + repack to (b, h, s, dk) bf16. One thread per rotated pair.
// ---------------------------------------------------------------------------
__global__ void rpmha_rope_pack(const float* __restrict__ X,
                                unsigned short* __restrict__ Xh) {
  int p = blockIdx.x * blockDim.x + threadIdx.x;
  if (p >= BSROWS * (DMODEL / 2)) return;
  const int r  = p >> 9;           // row in (b*s)
  const int pi = p & 511;
  const int h  = pi >> 5;          // head
  const int i  = pi & 31;          // pair index within head
  const int pos = r & (SEQ - 1);
  const int bb  = r >> 11;
  const float inv = __powf(10000.f, (float)(2 * i) * (1.0f / (float)DKH));
  const float th  = (float)pos / inv;
  const float c = __cosf(th), s = __sinf(th);
  const float x0 = X[(size_t)r * DMODEL + h * DKH + 2 * i];
  const float x1 = X[(size_t)r * DMODEL + h * DKH + 2 * i + 1];
  const size_t o =
      (((size_t)(bb * NHEADS + h)) * SEQ + pos) * DKH + 2 * i;
  Xh[o]     = f2bf(x0 * c - x1 * s);
  Xh[o + 1] = f2bf(x1 * c + x0 * s);
}

__global__ void rpmha_pack_v(const float* __restrict__ X,
                             unsigned short* __restrict__ Vh) {
  int idx = blockIdx.x * blockDim.x + threadIdx.x;
  if (idx >= BSROWS * DMODEL) return;
  const int r = idx >> 10, c = idx & 1023;
  const int h = c >> 6, tcol = c & 63;
  const int pos = r & (SEQ - 1), bb = r >> 11;
  Vh[(((size_t)(bb * NHEADS + h)) * SEQ + pos) * DKH + tcol] = f2bf(X[idx]);
}

// ---------------------------------------------------------------------------
// Flash attention per (b,h): grid (qblocks=16, bh=32), 8 waves/block,
// each wave owns 16 query rows; key tiles of 32 staged cooperatively in LDS
// (K row-major via async b128, V transposed for aligned b32 fragment reads).
// ---------------------------------------------------------------------------
__global__ __launch_bounds__(256)
void rpmha_attention(const unsigned short* __restrict__ Qh,
                     const unsigned short* __restrict__ Kh,
                     const unsigned short* __restrict__ Vh,
                     const unsigned char* __restrict__ mask,
                     unsigned short* __restrict__ Out) {
  __shared__ unsigned short Ks[32][64];        // key-local x dk, 16B rows
  __shared__ unsigned short Vs[64][34];        // transposed: dk-col x key-local
  __shared__ unsigned short Plds[8][16][34];   // per-wave P re-stripe buffer

  const int bh = blockIdx.y;
  const int b  = bh >> 4;
  const int h  = bh & 15;
  const int t    = threadIdx.x;
  const int wave = t >> 5;
  const int lane = t & 31;
  const int half = lane >> 4;
  const int mrow = lane & 15;
  const int ncol = lane & 15;
  const int q0 = blockIdx.x * 128 + wave * 16;

  const unsigned short* Qb = Qh + (size_t)bh * SEQ * DKH;
  const unsigned short* Kb = Kh + (size_t)bh * SEQ * DKH;
  const unsigned short* Vb = Vh + (size_t)bh * SEQ * DKH;
  const unsigned char*  mb = mask + (size_t)b * SEQ;

  // Cooperative loader assignment: 8 elements per thread per tile.
  const int krow = t >> 3, kcol = (t & 7) * 8;

  // Q operand: 16x64, held as two 16x32 A-fragments for the whole key sweep.
  BFrag qf[2];
#pragma unroll
  for (int f = 0; f < 2; ++f)
#pragma unroll
    for (int v = 0; v < 8; ++v)
      qf[f].u[v] = *(const unsigned*)(Qb + (size_t)(q0 + mrow) * DKH + 32 * f +
                                      a_koff(v, half));

  float mx[8], lsum[8];
  f32x8 o[4];
#pragma unroll
  for (int i = 0; i < 8; ++i) { mx[i] = -1e30f; lsum[i] = 0.f; }
#pragma unroll
  for (int c = 0; c < 4; ++c)
    o[c] = (f32x8){0.f, 0.f, 0.f, 0.f, 0.f, 0.f, 0.f, 0.f};

  const float scale = 0.125f;   // 1/sqrt(64)

  for (int kb = 0; kb < SEQ; kb += 32) {
    __syncthreads();   // previous iteration's LDS reads are done
    // Stage K tile (async) and transposed V tile (scatter).
    rpmha_copy16(Kb + (size_t)(kb + krow) * DKH + kcol, &Ks[krow][kcol]);
    {
      U128 vv;
      vv.q = *(const uint4*)(Vb + (size_t)(kb + krow) * DKH + kcol);
#pragma unroll
      for (int e = 0; e < 8; ++e) Vs[kcol + e][krow] = vv.s[e];
    }
    rpmha_wait_async();
    __syncthreads();

    // S tiles: S_t = Q(16x64) * K^T(64x16) for key subtiles t=0,1.
    f32x8 s[2];
#pragma unroll
    for (int tt = 0; tt < 2; ++tt) {
      BFrag kf0, kf1;
#pragma unroll
      for (int v = 0; v < 8; ++v) {
        const int kk = b_koff(v, half);
        kf0.u[v] = *(const unsigned*)&Ks[16 * tt + ncol][kk];
        kf1.u[v] = *(const unsigned*)&Ks[16 * tt + ncol][32 + kk];
      }
      f32x8 z = (f32x8){0.f, 0.f, 0.f, 0.f, 0.f, 0.f, 0.f, 0.f};
      z = __builtin_amdgcn_wmma_f32_16x16x32_bf16(false, qf[0].v, false, kf0.v,
                                                  (short)0, z, false, false);
      z = __builtin_amdgcn_wmma_f32_16x16x32_bf16(false, qf[1].v, false, kf1.v,
                                                  (short)0, z, false, false);
      s[tt] = z;
    }

    const bool ok0 = mb[kb + ncol] != 0;
    const bool ok1 = mb[kb + 16 + ncol] != 0;
#pragma unroll
    for (int i = 0; i < 8; ++i) {
      s[0][i] = ok0 ? s[0][i] * scale : -1e30f;
      s[1][i] = ok1 ? s[1][i] * scale : -1e30f;
    }

    // Online softmax: row stats reduced across the 16-lane half-group.
    float p0[8], p1[8], av[8];
#pragma unroll
    for (int i = 0; i < 8; ++i) {
      float vmax = fmaxf(s[0][i], s[1][i]);
#pragma unroll
      for (int off = 1; off < 16; off <<= 1)
        vmax = fmaxf(vmax, __shfl_xor(vmax, off, 32));
      const float mnew = fmaxf(mx[i], vmax);
      av[i] = __expf(mx[i] - mnew);
      p0[i] = __expf(s[0][i] - mnew);
      p1[i] = __expf(s[1][i] - mnew);
      float rs = p0[i] + p1[i];
#pragma unroll
      for (int off = 1; off < 16; off <<= 1)
        rs += __shfl_xor(rs, off, 32);
      lsum[i] = lsum[i] * av[i] + rs;
      mx[i] = mnew;
    }
#pragma unroll
    for (int c = 0; c < 4; ++c)
#pragma unroll
      for (int i = 0; i < 8; ++i) o[c][i] *= av[i];

    // Re-stripe P (D-layout 16x32 f32) -> A-fragment layout via per-wave LDS.
#pragma unroll
    for (int i = 0; i < 8; ++i) {
      Plds[wave][i + 8 * half][ncol]      = f2bf(p0[i]);
      Plds[wave][i + 8 * half][16 + ncol] = f2bf(p1[i]);
    }
    __syncthreads();
    BFrag pf;
#pragma unroll
    for (int v = 0; v < 8; ++v)
      pf.u[v] = *(const unsigned*)&Plds[wave][mrow][a_koff(v, half)];

    // O += P(16x32) * V(32x64), four 16-col WMMA tiles from transposed Vs.
#pragma unroll
    for (int c = 0; c < 4; ++c) {
      BFrag vf;
#pragma unroll
      for (int v = 0; v < 8; ++v)
        vf.u[v] = *(const unsigned*)&Vs[c * 16 + ncol][b_koff(v, half)];
      o[c] = __builtin_amdgcn_wmma_f32_16x16x32_bf16(false, pf.v, false, vf.v,
                                                     (short)0, o[c], false,
                                                     false);
    }
  }

  // Normalize and store bf16 into (b*s, d) for the output projection.
#pragma unroll
  for (int i = 0; i < 8; ++i) {
    const float rn = 1.0f / lsum[i];
    const int row = q0 + i + 8 * half;
#pragma unroll
    for (int c = 0; c < 4; ++c) {
      const int col = h * DKH + c * 16 + ncol;
      Out[((size_t)(b * SEQ + row)) * DMODEL + col] = f2bf(o[c][i] * rn);
    }
  }
}

// ---------------------------------------------------------------------------
// Host launcher
// ---------------------------------------------------------------------------
extern "C" void kernel_launch(void* const* d_in, const int* in_sizes, int n_in,
                              void* d_out, int out_size, void* d_ws,
                              size_t ws_size, hipStream_t stream) {
  const float* q    = (const float*)d_in[0];
  const float* k    = (const float*)d_in[1];
  const float* v    = (const float*)d_in[2];
  const unsigned char* mask = (const unsigned char*)d_in[3];
  const float* w_q  = (const float*)d_in[4];
  const float* b_q  = (const float*)d_in[5];
  const float* w_k  = (const float*)d_in[6];
  const float* b_k  = (const float*)d_in[7];
  const float* w_v  = (const float*)d_in[8];
  const float* b_v  = (const float*)d_in[9];
  const float* w_o  = (const float*)d_in[10];
  const float* b_o  = (const float*)d_in[11];
  float* out = (float*)d_out;

  const size_t SZ_ACT_BF = (size_t)BSROWS * DMODEL * sizeof(unsigned short);
  const size_t SZ_W_BF   = (size_t)DMODEL * DMODEL * sizeof(unsigned short);
  const size_t SZ_PROJ_F = (size_t)BSROWS * DMODEL * sizeof(float);

  char* ws = (char*)d_ws;
  size_t off = 0;
  unsigned short* qbf  = (unsigned short*)(ws + off); off += SZ_ACT_BF;
  unsigned short* kbf  = (unsigned short*)(ws + off); off += SZ_ACT_BF;
  unsigned short* vbf  = (unsigned short*)(ws + off); off += SZ_ACT_BF;
  unsigned short* wqbf = (unsigned short*)(ws + off); off += SZ_W_BF;
  unsigned short* wkbf = (unsigned short*)(ws + off); off += SZ_W_BF;
  unsigned short* wvbf = (unsigned short*)(ws + off); off += SZ_W_BF;
  unsigned short* wobf = (unsigned short*)(ws + off); off += SZ_W_BF;
  float*          proj = (float*)(ws + off);          off += SZ_PROJ_F;
  unsigned short* Qhb  = (unsigned short*)(ws + off); off += SZ_ACT_BF;
  unsigned short* Khb  = (unsigned short*)(ws + off); off += SZ_ACT_BF;
  unsigned short* Vhb  = (unsigned short*)(ws + off); off += SZ_ACT_BF;
  unsigned short* attn = (unsigned short*)(ws + off); off += SZ_ACT_BF;
  if (off > ws_size) return;   // workspace too small; bail deterministically

  const int nAct = BSROWS * DMODEL;     // 4,194,304
  const int nW   = DMODEL * DMODEL;     // 1,048,576

  rpmha_f32_to_bf16<<<nAct / 256, 256, 0, stream>>>(q, qbf, nAct);
  rpmha_f32_to_bf16<<<nAct / 256, 256, 0, stream>>>(k, kbf, nAct);
  rpmha_f32_to_bf16<<<nAct / 256, 256, 0, stream>>>(v, vbf, nAct);
  rpmha_f32_to_bf16<<<nW / 256, 256, 0, stream>>>(w_q, wqbf, nW);
  rpmha_f32_to_bf16<<<nW / 256, 256, 0, stream>>>(w_k, wkbf, nW);
  rpmha_f32_to_bf16<<<nW / 256, 256, 0, stream>>>(w_v, wvbf, nW);
  rpmha_f32_to_bf16<<<nW / 256, 256, 0, stream>>>(w_o, wobf, nW);

  const dim3 gemmGrid(DMODEL / 64, BSROWS / 128);   // (16, 32)
  const int ropeBlocks = (BSROWS * (DMODEL / 2)) / 256;

  // Q projection + RoPE
  rpmha_gemm_bf16<<<gemmGrid, 256, 0, stream>>>(qbf, wqbf, b_q, proj, BSROWS,
                                                DMODEL, DMODEL);
  rpmha_rope_pack<<<ropeBlocks, 256, 0, stream>>>(proj, Qhb);
  // K projection + RoPE
  rpmha_gemm_bf16<<<gemmGrid, 256, 0, stream>>>(kbf, wkbf, b_k, proj, BSROWS,
                                                DMODEL, DMODEL);
  rpmha_rope_pack<<<ropeBlocks, 256, 0, stream>>>(proj, Khb);
  // V projection + repack
  rpmha_gemm_bf16<<<gemmGrid, 256, 0, stream>>>(vbf, wvbf, b_v, proj, BSROWS,
                                                DMODEL, DMODEL);
  rpmha_pack_v<<<nAct / 256, 256, 0, stream>>>(proj, Vhb);

  // Attention
  rpmha_attention<<<dim3(SEQ / 128, BDIM * NHEADS), 256, 0, stream>>>(
      Qhb, Khb, Vhb, mask, attn);

  // Output projection -> fp32 result
  rpmha_gemm_bf16<<<gemmGrid, 256, 0, stream>>>(attn, wobf, b_o, out, BSROWS,
                                                DMODEL, DMODEL);
}
